// NBP_oc_3977139716402
// MI455X (gfx1250) — compile-verified, matrix operand order checked by reading, not run on hardware
//
#include <hip/hip_runtime.h>
#include <hip/hip_bf16.h>

// Problem constants (from reference): B=120, N=128, M=192, T=25
#define BDIM 120
#define NDIM 128
#define MDIM 192
#define TITER 25
#define KMAX 16   // max nonzeros per H row (mean 4, Binom(128,1/32); 16 is ~8 sigma)
#define CMAX 32   // max nonzeros per H column (mean 6)

typedef __attribute__((ext_vector_type(16))) _Float16 v16h;
typedef __attribute__((ext_vector_type(8)))  float    v8f;

// phi(x) = -log(tanh(x/2)) = log((e^x + 1)/(e^x - 1)),  x clipped to [1e-6, 30]
__device__ __forceinline__ float phi_f(float x) {
    x = fminf(fmaxf(x, 1e-6f), 30.0f);
    float ex = __expf(x);
    return __logf((ex + 1.0f) / (ex - 1.0f));
}

// ---------------------------------------------------------------------------
// Build CSR-like row lists: for each (s,m), the column indices n with Hs>0.
// s=0 -> Hz, s=1 -> Hx (matches Hs = stack([Hz, Hx])).
// ---------------------------------------------------------------------------
__global__ void prep_rows(const float* __restrict__ Hx, const float* __restrict__ Hz,
                          int* __restrict__ row_cnt, int* __restrict__ row_col) {
    int r = blockIdx.x * blockDim.x + threadIdx.x;
    if (r >= 2 * MDIM) return;
    int s = r / MDIM, m = r - s * MDIM;
    const float* H = (s == 0) ? Hz : Hx;
    int cnt = 0;
    for (int n = 0; n < NDIM; ++n) {
        if (H[m * NDIM + n] > 0.5f && cnt < KMAX) {
            row_col[r * KMAX + cnt] = n;
            ++cnt;
        }
    }
    row_cnt[r] = cnt;
}

// Column lists: for each (s,n), flat indices (s*M+m)*KMAX+k of support entries,
// scanned in ascending m -> deterministic summation order.
__global__ void prep_cols(const int* __restrict__ row_cnt, const int* __restrict__ row_col,
                          int* __restrict__ col_cnt, int* __restrict__ col_ptr) {
    int c = blockIdx.x * blockDim.x + threadIdx.x;
    if (c >= 2 * NDIM) return;
    int s = c >> 7, n = c & (NDIM - 1);
    int cnt = 0;
    for (int m = 0; m < MDIM; ++m) {
        int r = s * MDIM + m;
        int rc = row_cnt[r];
        for (int k = 0; k < rc; ++k) {
            if (row_col[r * KMAX + k] == n && cnt < CMAX) {
                col_ptr[c * CMAX + cnt] = r * KMAX + k;
                ++cnt;
            }
        }
    }
    col_cnt[c] = cnt;
}

// ---------------------------------------------------------------------------
// Syndromes via WMMA:  C[b,m] = sum_n E[b,n] * H[m,n]   (exact in f16: 0/1 data,
// sums <= 128).  syn_sign[b,s,m] = 1 - 2*(C mod 2).
// s=0: (errorx, Hz) ; s=1: (errorz, Hx).
// One wave per 16x16 output tile; K = 128 in 4 steps of 32.
//
// Out-of-range batch rows (120..127, tile 7 only) need no zeroing: in a WMMA,
// D row b depends only on A row b, and those D rows are masked at the store.
// We just clamp the address -> unconditional float4 (b128) loads, no branches.
// ---------------------------------------------------------------------------
__device__ __forceinline__ void pack8(v16h& dst, int base, float4 lo, float4 hi) {
    dst[base + 0] = (_Float16)lo.x;  dst[base + 1] = (_Float16)lo.y;
    dst[base + 2] = (_Float16)lo.z;  dst[base + 3] = (_Float16)lo.w;
    dst[base + 4] = (_Float16)hi.x;  dst[base + 5] = (_Float16)hi.y;
    dst[base + 6] = (_Float16)hi.z;  dst[base + 7] = (_Float16)hi.w;
}

__global__ void syndrome_wmma(const float* __restrict__ errx,
                              const float* __restrict__ errz,
                              const float* __restrict__ Hx,
                              const float* __restrict__ Hz,
                              float* __restrict__ syn_sign) {
    const int tb = blockIdx.x;      // batch tile 0..7 (rows, padded 120->128)
    const int tm = blockIdx.y;      // M tile 0..11
    const int s  = blockIdx.z;      // code select
    const int lane = threadIdx.x;   // wave32
    const float* E = s ? errz : errx;
    const float* H = s ? Hx  : Hz;

    const int half = lane >> 4;
    const int lr   = lane & 15;
    const int arow = tb * 16 + lr;                       // batch row (A)
    const int arow_c = (arow < BDIM) ? arow : (BDIM - 1); // clamp addr only
    const int bcol = tm * 16 + lr;                       // M column (B = H row)

    const float4* Ev = (const float4*)(E + arow_c * NDIM);
    const float4* Hv = (const float4*)(H + bcol * NDIM);

    v8f c = {};
    #pragma unroll
    for (int kk = 0; kk < 4; ++kk) {
        // ISA 16-bit A 16x32 layout: lanes 0-15 hold K {0..7,16..23},
        // lanes 16-31 hold K {8..15,24..31}; B mirrors with N in place of M.
        const int q = 8 * kk + 2 * half;   // float4 index of K block start
        const float4 a0 = Ev[q],     a1 = Ev[q + 1];
        const float4 a2 = Ev[q + 4], a3 = Ev[q + 5];
        const float4 b0 = Hv[q],     b1 = Hv[q + 1];
        const float4 b2 = Hv[q + 4], b3 = Hv[q + 5];
        v16h a, bm;
        pack8(a,  0, a0, a1);
        pack8(a,  8, a2, a3);
        pack8(bm, 0, b0, b1);
        pack8(bm, 8, b2, b3);
        c = __builtin_amdgcn_wmma_f32_16x16x32_f16(false, a, false, bm,
                                                   (short)0, c, false, false);
    }
    // C/D f32 layout: VGPR r, lanes 0-15 -> M=r, lanes 16-31 -> M=r+8; N=lane&15
    #pragma unroll
    for (int r = 0; r < 8; ++r) {
        const int brow = tb * 16 + r + half * 8;
        const int mcol = tm * 16 + lr;
        if (brow < BDIM) {
            const int par = __float2int_rn(c[r]) & 1;
            syn_sign[(brow * 2 + s) * MDIM + mcol] = 1.0f - 2.0f * (float)par;
        }
    }
}

// ---------------------------------------------------------------------------
// Main BP loop: one workgroup per batch element. 384 threads = 12 waves.
// Thread r (<384) owns check-node row (s,m): private KMAX slots in Vsh/Csh.
// Thread c (<256) owns variable column (s,n): deterministic ordered column sum.
// ---------------------------------------------------------------------------
__global__ __launch_bounds__(384, 1)
void nbp_main(const float* __restrict__ errx, const float* __restrict__ errz,
              const float* __restrict__ ep0,
              const float* __restrict__ Gx, const float* __restrict__ Gz,
              const float* __restrict__ wllr_a, const float* __restrict__ wvn_a,
              const float* __restrict__ wcn_a,
              const float* __restrict__ syn_sign,
              const int* __restrict__ row_cnt, const int* __restrict__ row_col,
              const int* __restrict__ col_cnt, const int* __restrict__ col_ptr,
              float* __restrict__ out_sum, float* __restrict__ out_min) {
    __shared__ float Vsh[2 * MDIM * KMAX];     // 24 KB  messages V (support only)
    __shared__ float Csh[2 * MDIM * KMAX];     // 24 KB  p-values then cn messages
    __shared__ float cn_sum[2 * NDIM];         // 1 KB
    __shared__ float corr[2 * NDIM];           // [0..127]=corrz, [128..255]=corrx
    __shared__ float ssign[2 * MDIM];          // syn_sign for this batch
    __shared__ float prodabs[4];
    __shared__ float acc[2];                   // {loss_sum, loss_min}

    const int b   = blockIdx.x;
    const int tid = threadIdx.x;

    const float ep   = ep0[0];
    const float llr0 = __logf((1.0f - 2.0f * ep / 3.0f) / (2.0f * ep / 3.0f));

    for (int i = tid; i < 2 * MDIM; i += blockDim.x)
        ssign[i] = syn_sign[b * 2 * MDIM + i];
    for (int i = tid; i < 2 * MDIM * KMAX; i += blockDim.x)
        Vsh[i] = llr0;                          // V0 = llr0 * Hs on support
    if (tid == 0) { acc[0] = 0.0f; acc[1] = 3.0e38f; }
    __syncthreads();

    // per-thread row metadata (thread-private LDS slots -> no cross-thread hazards)
    const int s_r   = tid / MDIM;
    const int rbase = tid * KMAX;
    const int rcnt  = row_cnt[tid];
    int cols[KMAX];
    #pragma unroll
    for (int k = 0; k < KMAX; ++k)
        cols[k] = (k < rcnt) ? row_col[rbase + k] : 0;

    // per-thread column metadata
    const int s_c  = tid >> 7;
    const int n_c  = tid & (NDIM - 1);
    const int ccnt = (tid < 2 * NDIM) ? col_cnt[tid] : 0;
    const int* cp  = &col_ptr[tid * CMAX];

    for (int t = 0; t < TITER; ++t) {
        const float w_llr = wllr_a[t];
        const float w_vn  = wvn_a[t];
        const float w_cn  = wcn_a[t];

        // ---- check-node (row) pass: signs, phi, leave-one-out phi -> cn
        {
            float psum = 0.0f, sprod = 1.0f;
            for (int k = 0; k < rcnt; ++k) {
                float v = Vsh[rbase + k];
                sprod = (v < 0.0f) ? -sprod : sprod;
                float p = phi_f(fabsf(v));
                Csh[rbase + k] = p;
                psum += p;
            }
            const float c0 = ssign[tid] * sprod * w_cn;
            for (int k = 0; k < rcnt; ++k) {
                float v  = Vsh[rbase + k];
                float sg = (v < 0.0f) ? -1.0f : 1.0f;
                float p  = Csh[rbase + k];
                Csh[rbase + k] = c0 * sg * phi_f(psum - p);
            }
        }
        __syncthreads();

        // ---- variable-node (column) sum + Gamma + correlation terms
        if (tid < 2 * NDIM) {
            float sum = 0.0f;
            for (int j = 0; j < ccnt; ++j) sum += Csh[cp[j]];  // fixed m-order
            cn_sum[tid] = sum;
            const float gamma = w_llr * llr0 + sum;
            const float prob  = 1.0f / (1.0f + __expf(gamma)); // sigmoid(-gamma)
            // s=0 (pX) pairs with errorz -> corrz ; s=1 (pZ) pairs with errorx -> corrx
            const float e = (s_c == 0) ? errz[b * NDIM + n_c] : errx[b * NDIM + n_c];
            corr[tid] = (e > 0.5f) ? (1.0f - prob) : prob;
        }
        __syncthreads();

        // ---- V update (thread-private slots)
        for (int k = 0; k < rcnt; ++k) {
            Vsh[rbase + k] = w_llr * llr0 +
                             w_vn * (cn_sum[s_r * NDIM + cols[k]] - Csh[rbase + k]);
        }
        // ---- loss dot products: waves 0,1 -> Gz·corrx ; waves 2,3 -> Gx·corrz
        if (tid < 128) {
            const int w    = tid >> 5;
            const int lane = tid & 31;
            const float* Grow = (w < 2) ? &Gz[w * NDIM] : &Gx[(w - 2) * NDIM];
            const float* cr   = (w < 2) ? &corr[NDIM]   : &corr[0];
            float part = 0.0f;
            for (int n = lane; n < NDIM; n += 32)
                part += Grow[n] * cr[n];
            #pragma unroll
            for (int off = 16; off > 0; off >>= 1)
                part += __shfl_down(part, off, 32);
            if (lane == 0)
                prodabs[w] = fabsf(__sinf(1.57079632679f * part));
        }
        __syncthreads();
        if (tid == 0) {
            const float li = prodabs[0] + prodabs[1] + prodabs[2] + prodabs[3];
            acc[0] += li;
            acc[1] = fminf(acc[1], li);
        }
        // next iteration's shared writes (cn_sum/corr/prodabs) only happen after
        // two more barriers, so thread 0's reads above are safe without a 4th.
    }
    if (tid == 0) { out_sum[b] = acc[0]; out_min[b] = acc[1]; }
}

// Deterministic final mean over the 120 batch elements -> out[0], out[1]
__global__ void final_reduce(const float* __restrict__ bsum,
                             const float* __restrict__ bmin,
                             float* __restrict__ out) {
    if (threadIdx.x == 0 && blockIdx.x == 0) {
        float s = 0.0f, m = 0.0f;
        for (int b = 0; b < BDIM; ++b) { s += bsum[b]; m += bmin[b]; }
        out[0] = s * (1.0f / (float)BDIM);
        out[1] = m * (1.0f / (float)BDIM);
    }
}

extern "C" void kernel_launch(void* const* d_in, const int* in_sizes, int n_in,
                              void* d_out, int out_size, void* d_ws, size_t ws_size,
                              hipStream_t stream) {
    // setup_inputs() order:
    const float* errx = (const float*)d_in[0];
    const float* errz = (const float*)d_in[1];
    const float* ep0  = (const float*)d_in[2];
    const float* Hx   = (const float*)d_in[3];
    const float* Hz   = (const float*)d_in[4];
    const float* Gx   = (const float*)d_in[5];
    const float* Gz   = (const float*)d_in[6];
    const float* wllr = (const float*)d_in[7];
    const float* wvn  = (const float*)d_in[8];
    const float* wcn  = (const float*)d_in[9];

    // workspace layout (all 4-byte elements, ~245 KB total)
    float* ws       = (float*)d_ws;
    float* syn_sign = ws;                                  // B*2*M = 46080
    int*   row_cnt  = (int*)(ws + BDIM * 2 * MDIM);        // 384
    int*   row_col  = row_cnt + 2 * MDIM;                  // 384*KMAX
    int*   col_cnt  = row_col + 2 * MDIM * KMAX;           // 256
    int*   col_ptr  = col_cnt + 2 * NDIM;                  // 256*CMAX
    float* bsum     = (float*)(col_ptr + 2 * NDIM * CMAX); // 120
    float* bmin     = bsum + BDIM;                         // 120

    hipLaunchKernelGGL(prep_rows, dim3(1), dim3(2 * MDIM), 0, stream,
                       Hx, Hz, row_cnt, row_col);
    hipLaunchKernelGGL(prep_cols, dim3(1), dim3(2 * NDIM), 0, stream,
                       row_cnt, row_col, col_cnt, col_ptr);
    hipLaunchKernelGGL(syndrome_wmma, dim3(8, 12, 2), dim3(32), 0, stream,
                       errx, errz, Hx, Hz, syn_sign);
    hipLaunchKernelGGL(nbp_main, dim3(BDIM), dim3(384), 0, stream,
                       errx, errz, ep0, Gx, Gz, wllr, wvn, wcn,
                       syn_sign, row_cnt, row_col, col_cnt, col_ptr, bsum, bmin);
    hipLaunchKernelGGL(final_reduce, dim3(1), dim3(32), 0, stream,
                       bsum, bmin, (float*)d_out);
}
